// CGDN_9964324127084
// MI455X (gfx1250) — matrix operand order repaired and checked by reading, not compiled, not run on hardware
//
#include <hip/hip_runtime.h>
#include <hip/hip_bf16.h>
#include <math.h>

#define HID 128
#define HEADS 4
#define CPH 32
#define NEG_SLOPE 0.2f
#define MP_SCALE 1000000.0f
#define MAX_DISP 50.0f

typedef __attribute__((ext_vector_type(16))) _Float16 v16h;
typedef __attribute__((ext_vector_type(8)))  _Float16 h8v;
typedef __attribute__((ext_vector_type(8)))  float    v8f;

__device__ __forceinline__ float gelu_f(float x) {
    // exact erf gelu, matches torch/jax approximate=False
    return 0.5f * x * (1.0f + erff(x * 0.70710678118654752440f));
}

// float atomic max via signed/unsigned bit trick (valid for mixed signs, -inf init)
__device__ __forceinline__ void atomic_max_f32(float* addr, float val) {
    if (val >= 0.0f) atomicMax((int*)addr, __float_as_int(val));
    else             atomicMin((unsigned int*)addr, (unsigned int)__float_as_int(val));
}

// ---------------- encoder: h = gelu(LN(x @ W + b)) ----------------
__global__ void encoder_kernel(const float* __restrict__ x,
                               const float* __restrict__ W,
                               const float* __restrict__ b,
                               const float* __restrict__ lng,
                               const float* __restrict__ lnb,
                               float* __restrict__ h, _Float16* __restrict__ h16, int N) {
    int n = blockIdx.x; int c = threadIdx.x;   // 128 threads / node
    float acc = b[c];
    #pragma unroll
    for (int j = 0; j < 7; ++j) acc += x[(size_t)n*7 + j] * W[j*HID + c];
    __shared__ float red[HID]; __shared__ float mu_s, var_s;
    red[c] = acc; __syncthreads();
    for (int s = 64; s > 0; s >>= 1) { if (c < s) red[c] += red[c+s]; __syncthreads(); }
    if (c == 0) mu_s = red[0] * (1.0f/HID);
    __syncthreads();
    float dv = acc - mu_s;
    red[c] = dv*dv; __syncthreads();
    for (int s = 64; s > 0; s >>= 1) { if (c < s) red[c] += red[c+s]; __syncthreads(); }
    if (c == 0) var_s = red[0] * (1.0f/HID);
    __syncthreads();
    float ln = dv * rsqrtf(var_s + 1e-5f) * lng[c] + lnb[c];
    float hv = gelu_f(ln);
    h[(size_t)n*HID + c]   = hv;
    h16[(size_t)n*HID + c] = (_Float16)hv;
}

// ---------------- FiLM: gb[n,0:128]=1+dgamma, gb[n,128:256]=beta ----------------
__global__ void film_kernel(const float* __restrict__ tmp,
                            const float* __restrict__ W1, const float* __restrict__ b1,
                            const float* __restrict__ W2, const float* __restrict__ b2,
                            float* __restrict__ gb, int N) {
    int n = blockIdx.x; int k = threadIdx.x;   // 256 threads / node
    __shared__ float hdn[64];
    float t = tmp[n] * (1.0f/MP_SCALE);
    if (k < 64) hdn[k] = gelu_f(t * W1[k] + b1[k]);
    __syncthreads();
    float acc = b2[k];
    #pragma unroll 8
    for (int j = 0; j < 64; ++j) acc += hdn[j] * W2[j*256 + k];
    gb[(size_t)n*256 + k] = (k < 128) ? (1.0f + acc) : acc;
}

// ---------------- pack f32 weight [K x Ncols] into WMMA f16 B-matrix layout ----------------
// Bp[((kk*NT+nt)*32+lane)*16+j] = W[kk*32 + (j<8? 8*(lane>>4)+j : 16+8*(lane>>4)+j-8)][nt*16 + (lane&15)]
__global__ void pack_b_kernel(const float* __restrict__ W, _Float16* __restrict__ Bp,
                              int Ncols, int total) {
    int i = blockIdx.x*blockDim.x + threadIdx.x;
    if (i >= total) return;
    int j    = i & 15;
    int lane = (i >> 4) & 31;
    int t2   = i >> 9;               // kk*NT + nt
    int NT   = Ncols >> 4;
    int nt = t2 % NT, kk = t2 / NT;
    int klocal = (j < 8) ? (8*(lane >> 4) + j) : (16 + 8*(lane >> 4) + (j - 8));
    int ncol   = nt*16 + (lane & 15);
    Bp[i] = (_Float16)W[(size_t)(kk*32 + klocal)*Ncols + ncol];
}

// ---------------- WMMA GEMM: C[M x NT*16] = A16[M x KT*32] @ B + bias (+gelu) ----------------
// one wave -> 16 rows x full Ncols; block = 8 waves = 128 rows
template<int KT, int NT, bool ACT>
__global__ void wmma_gemm_kernel(const _Float16* __restrict__ A,
                                 const _Float16* __restrict__ Bp,
                                 const float* __restrict__ bias,
                                 float* __restrict__ C, int M) {
    const int K = KT*32, Ncols = NT*16;
    int lane = threadIdx.x & 31;
    int wave = threadIdx.x >> 5;
    int row0 = blockIdx.x * 128 + wave * 16;
    if (row0 >= M) return;                       // whole-wave exit, EXEC stays full
    int ar   = row0 + (lane & 15);
    int aoff = (lane < 16) ? 0 : 8;
    const _Float16* abase = A + (size_t)ar * K + aoff;
    v8f zero = {};
    v8f acc[NT];
    #pragma unroll
    for (int nt = 0; nt < NT; ++nt) acc[nt] = zero;
    #pragma unroll
    for (int kk = 0; kk < KT; ++kk) {
        union { v16h v; h8v h[2]; } a;
        a.h[0] = *(const h8v*)(abase + kk*32);        // K = base+0..7   (or 8..15)
        a.h[1] = *(const h8v*)(abase + kk*32 + 16);   // K = base+16..23 (or 24..31)
        #pragma unroll
        for (int nt = 0; nt < NT; ++nt) {
            v16h b = *(const v16h*)(Bp + (size_t)((kk*NT + nt)*32 + lane)*16);
            acc[nt] = __builtin_amdgcn_wmma_f32_16x16x32_f16(
                false, a.v, false, b, (short)0, acc[nt], false, false);
        }
    }
    int colb  = lane & 15;
    int rplus = (lane < 16) ? 0 : 8;
    #pragma unroll
    for (int nt = 0; nt < NT; ++nt) {
        #pragma unroll
        for (int r = 0; r < 8; ++r) {
            int row = row0 + rplus + r;
            int col = nt*16 + colb;
            float v = acc[nt][r] + bias[col];
            if (ACT) v = gelu_f(v);
            C[(size_t)row*Ncols + col] = v;
        }
    }
}

// ---------------- per-layer init ----------------
__global__ void init_layer_kernel(float* __restrict__ acc, float* __restrict__ smax,
                                  float* __restrict__ den, int N) {
    int i = blockIdx.x*blockDim.x + threadIdx.x;
    if (i < N*HID) acc[i] = 0.0f;
    if (i < N*HEADS) { smax[i] = -__builtin_inff(); den[i] = 0.0f; }
}

// ---------------- edge pass 1: scores + segment max (wave per (edge,head)) ----------------
__global__ void edge_score_kernel(const int* __restrict__ src, const int* __restrict__ dst,
                                  const float* __restrict__ ea, const float* __restrict__ We,
                                  const float* __restrict__ att,
                                  const float* __restrict__ xl, const float* __restrict__ xr,
                                  float* __restrict__ score, float* __restrict__ smax, int E) {
    int gid = blockIdx.x*blockDim.x + threadIdx.x;
    int w = gid >> 5, lane = gid & 31;
    if (w >= E*HEADS) return;
    int e = w >> 2, hd = w & 3;
    int s = src[e], d = dst[e];
    int ch = hd*CPH + lane;
    float ev = ea[(size_t)e*4+0]*We[0*HID+ch] + ea[(size_t)e*4+1]*We[1*HID+ch]
             + ea[(size_t)e*4+2]*We[2*HID+ch] + ea[(size_t)e*4+3]*We[3*HID+ch];
    float m = xl[(size_t)s*HID + ch] + xr[(size_t)d*HID + ch] + ev;
    m = (m > 0.0f) ? m : NEG_SLOPE*m;
    float v = m * att[ch];
    #pragma unroll
    for (int o = 16; o > 0; o >>= 1) v += __shfl_xor(v, o, 32);
    if (lane == 0) {
        score[(size_t)e*HEADS + hd] = v;
        atomic_max_f32(&smax[(size_t)d*HEADS + hd], v);
    }
}

// ---------------- edge pass 2: exp + segment sum (thread per (edge,head)) ----------------
__global__ void edge_exp_kernel(const int* __restrict__ dst, float* __restrict__ score,
                                const float* __restrict__ smax, float* __restrict__ den, int E) {
    int i = blockIdx.x*blockDim.x + threadIdx.x;
    if (i >= E*HEADS) return;
    int e = i >> 2, hd = i & 3;
    int d = dst[e];
    float ex = expf(score[i] - smax[(size_t)d*HEADS + hd]);
    score[i] = ex;                                    // overwrite score with exp
    atomicAdd(&den[(size_t)d*HEADS + hd], ex);
}

// ---------------- edge pass 3: alpha-weighted scatter of source features ----------------
__global__ void edge_aggr_kernel(const int* __restrict__ src, const int* __restrict__ dst,
                                 const float* __restrict__ score, const float* __restrict__ den,
                                 const float* __restrict__ xl, float* __restrict__ acc, int E) {
    int gid = blockIdx.x*blockDim.x + threadIdx.x;
    int w = gid >> 5, lane = gid & 31;
    if (w >= E*HEADS) return;
    int e = w >> 2, hd = w & 3;
    int s = src[e], d = dst[e];
    float alpha = score[(size_t)e*HEADS + hd] / (den[(size_t)d*HEADS + hd] + 1e-16f);
    int ch = hd*CPH + lane;
    atomicAdd(&acc[(size_t)d*HID + ch], alpha * xl[(size_t)s*HID + ch]);
}

// ---------------- layer epilogue: bias + LN + FiLM + gelu + residual ----------------
__global__ void layer_epilogue_kernel(const float* __restrict__ accb,
                                      const float* __restrict__ gbias,
                                      const float* __restrict__ lng, const float* __restrict__ lnb,
                                      const float* __restrict__ gb,
                                      float* __restrict__ h, _Float16* __restrict__ h16, int N) {
    int n = blockIdx.x; int c = threadIdx.x;   // 128 threads / node
    float v = accb[(size_t)n*HID + c] + gbias[c];
    __shared__ float red[HID]; __shared__ float mu_s, var_s;
    red[c] = v; __syncthreads();
    for (int s = 64; s > 0; s >>= 1) { if (c < s) red[c] += red[c+s]; __syncthreads(); }
    if (c == 0) mu_s = red[0] * (1.0f/HID);
    __syncthreads();
    float dv = v - mu_s;
    red[c] = dv*dv; __syncthreads();
    for (int s = 64; s > 0; s >>= 1) { if (c < s) red[c] += red[c+s]; __syncthreads(); }
    if (c == 0) var_s = red[0] * (1.0f/HID);
    __syncthreads();
    float ln = dv * rsqrtf(var_s + 1e-5f) * lng[c] + lnb[c];
    float f  = gb[(size_t)n*256 + c] * ln + gb[(size_t)n*256 + 128 + c];
    float hn = gelu_f(f) + h[(size_t)n*HID + c];      // residual = pre-layer h
    h[(size_t)n*HID + c]   = hn;
    h16[(size_t)n*HID + c] = (_Float16)hn;
}

// ---------------- decoder tail: delta = clip(d @ W2 + b2) * free; out = (coords+delta, delta) ----
__global__ void final_kernel(const float* __restrict__ x, const float* __restrict__ dbuf,
                             const float* __restrict__ W2, const float* __restrict__ b2,
                             const unsigned char* __restrict__ fixedm,
                             float* __restrict__ out, int N) {
    int n = blockIdx.x*blockDim.x + threadIdx.x;
    if (n >= N) return;
    float d0 = b2[0], d1 = b2[1];
    const float* dr = dbuf + (size_t)n*64;
    #pragma unroll 8
    for (int j = 0; j < 64; ++j) { float dj = dr[j]; d0 += dj*W2[j*2+0]; d1 += dj*W2[j*2+1]; }
    d0 = fminf(fmaxf(d0, -MAX_DISP), MAX_DISP);
    d1 = fminf(fmaxf(d1, -MAX_DISP), MAX_DISP);
    float fm = fixedm[n] ? 0.0f : 1.0f;
    d0 *= fm; d1 *= fm;
    out[(size_t)n*2+0] = x[(size_t)n*7+0] + d0;
    out[(size_t)n*2+1] = x[(size_t)n*7+1] + d1;
    out[(size_t)2*N + n*2+0] = d0;
    out[(size_t)2*N + n*2+1] = d1;
}

extern "C" void kernel_launch(void* const* d_in, const int* in_sizes, int n_in,
                              void* d_out, int out_size, void* d_ws, size_t ws_size,
                              hipStream_t stream) {
    (void)n_in; (void)out_size; (void)ws_size;
    const int N = in_sizes[0] / 7;
    const int E = in_sizes[1] / 2;

    const float* x   = (const float*)d_in[0];
    const int*   ei  = (const int*)d_in[1];
    const int*   src = ei;
    const int*   dst = ei + E;
    const float* ea  = (const float*)d_in[2];
    const float* tmp = (const float*)d_in[3];
    const unsigned char* fixedm = (const unsigned char*)d_in[4];
    const float* encW  = (const float*)d_in[5];
    const float* encB  = (const float*)d_in[6];
    const float* encG  = (const float*)d_in[7];
    const float* encLb = (const float*)d_in[8];
    struct Layer { const float *W1,*b1,*W2,*b2,*Wl,*bl,*Wr,*br,*We,*att,*bias,*lng,*lnb; } lay[4];
    for (int i = 0; i < 4; ++i) {
        int b_ = 9 + 13*i;
        lay[i].W1  = (const float*)d_in[b_+0];  lay[i].b1  = (const float*)d_in[b_+1];
        lay[i].W2  = (const float*)d_in[b_+2];  lay[i].b2  = (const float*)d_in[b_+3];
        lay[i].Wl  = (const float*)d_in[b_+4];  lay[i].bl  = (const float*)d_in[b_+5];
        lay[i].Wr  = (const float*)d_in[b_+6];  lay[i].br  = (const float*)d_in[b_+7];
        lay[i].We  = (const float*)d_in[b_+8];  lay[i].att = (const float*)d_in[b_+9];
        lay[i].bias= (const float*)d_in[b_+10];
        lay[i].lng = (const float*)d_in[b_+11]; lay[i].lnb = (const float*)d_in[b_+12];
    }
    const float* decW1 = (const float*)d_in[61];
    const float* decB1 = (const float*)d_in[62];
    const float* decW2 = (const float*)d_in[63];
    const float* decB2 = (const float*)d_in[64];

    // ---- workspace carve-up (256B aligned) ----
    char* p = (char*)d_ws;
    auto alloc = [&](size_t bytes) { char* r = p; p += (bytes + 255) & ~(size_t)255; return r; };
    float*    h     = (float*)alloc((size_t)N*HID*4);
    float*    accb  = (float*)alloc((size_t)N*HID*4);
    float*    xl    = (float*)alloc((size_t)N*HID*4);
    float*    xr    = (float*)alloc((size_t)N*HID*4);
    float*    gb    = (float*)alloc((size_t)N*256*4);
    float*    score = (float*)alloc((size_t)E*HEADS*4);
    float*    smax  = (float*)alloc((size_t)N*HEADS*4);
    float*    den   = (float*)alloc((size_t)N*HEADS*4);
    _Float16* h16   = (_Float16*)alloc((size_t)N*HID*2);
    _Float16* bpL[4]; _Float16* bpR[4];
    for (int i = 0; i < 4; ++i) {
        bpL[i] = (_Float16*)alloc((size_t)HID*HID*2);
        bpR[i] = (_Float16*)alloc((size_t)HID*HID*2);
    }
    _Float16* bpDec = (_Float16*)alloc((size_t)HID*64*2);
    float* dbuf = xl;   // decoder hidden reuses xl after last layer

    // ---- encoder + weight packing ----
    encoder_kernel<<<N, HID, 0, stream>>>(x, encW, encB, encG, encLb, h, h16, N);
    for (int i = 0; i < 4; ++i) {
        pack_b_kernel<<<(HID*HID+255)/256, 256, 0, stream>>>(lay[i].Wl, bpL[i], HID, HID*HID);
        pack_b_kernel<<<(HID*HID+255)/256, 256, 0, stream>>>(lay[i].Wr, bpR[i], HID, HID*HID);
    }
    pack_b_kernel<<<(HID*64+255)/256, 256, 0, stream>>>(decW1, bpDec, 64, HID*64);

    const int gemm_grid   = (N + 127) / 128;
    const long ewaves     = (long)E * HEADS;
    const int escore_grid = (int)((ewaves + 7) / 8);     // 8 waves / 256-thread block
    const int ethr_grid   = (int)((ewaves + 255) / 256);
    const int init_grid   = (N*HID + 255) / 256;

    for (int i = 0; i < 4; ++i) {
        film_kernel<<<N, 256, 0, stream>>>(tmp, lay[i].W1, lay[i].b1, lay[i].W2, lay[i].b2, gb, N);
        wmma_gemm_kernel<4,8,false><<<gemm_grid, 256, 0, stream>>>(h16, bpL[i], lay[i].bl, xl, N);
        wmma_gemm_kernel<4,8,false><<<gemm_grid, 256, 0, stream>>>(h16, bpR[i], lay[i].br, xr, N);
        init_layer_kernel<<<init_grid, 256, 0, stream>>>(accb, smax, den, N);
        edge_score_kernel<<<escore_grid, 256, 0, stream>>>(src, dst, ea, lay[i].We, lay[i].att,
                                                           xl, xr, score, smax, E);
        edge_exp_kernel<<<ethr_grid, 256, 0, stream>>>(dst, score, smax, den, E);
        edge_aggr_kernel<<<escore_grid, 256, 0, stream>>>(src, dst, score, den, xl, accb, E);
        layer_epilogue_kernel<<<N, HID, 0, stream>>>(accb, lay[i].bias, lay[i].lng, lay[i].lnb,
                                                     gb, h, h16, N);
    }
    wmma_gemm_kernel<4,4,true><<<gemm_grid, 256, 0, stream>>>(h16, bpDec, decB1, dbuf, N);
    final_kernel<<<(N+255)/256, 256, 0, stream>>>(x, dbuf, decW2, decB2, fixedm, (float*)d_out, N);
}
